// SRNN_custom_6038724018571
// MI455X (gfx1250) — compile-verified, matrix operand order checked by reading, not run on hardware
//
#include <hip/hip_runtime.h>
#include <hip/hip_bf16.h>

// ---------------------------------------------------------------------------
// SRNN on MI455X (gfx1250, wave32, WMMA).
//   Phase 1: x @ W_ih1^T as bf16 hi/lo-split WMMA GEMM (HBM-roofline bound).
//   Phase 2: persistent per-batch-tile scan; spikes are {0,1} => bf16-exact,
//            weights split into bf16 hi+lo => recurrent GEMMs are fp32-exact.
// ---------------------------------------------------------------------------

#define BATCH  512
#define TSTEPS 250
#define DIN    700
#define HID    128
#define OUTN   20
#define B_J0   0.01f
#define BETA   1.8f

typedef __attribute__((ext_vector_type(16))) __bf16 bf16x16;
typedef __attribute__((ext_vector_type(8)))  __bf16 bf16x8;
typedef __attribute__((ext_vector_type(8)))  float  v8f;

#define WMMA_BF16(A, B, C) \
  __builtin_amdgcn_wmma_f32_16x16x32_bf16(false, (A), false, (B), (short)0, (C), false, false)

// ws layout (bf16 elements): swizzled B-operand weights, [kt][nt][lane][16]
#define CNT_IH (22 * 8 * 32 * 16)  // W_ih1^T, K padded 700->704
#define CNT_R  (4 * 8 * 32 * 16)   // 128x128 recurrent matrices
#define CNT_O  (4 * 2 * 32 * 16)   // W_h2o^T, N padded 20->32
#define OFF_IH_HI ((size_t)0)
#define OFF_IH_LO (OFF_IH_HI + CNT_IH)
#define OFF_11_HI (OFF_IH_LO + CNT_IH)
#define OFF_11_LO (OFF_11_HI + CNT_R)
#define OFF_12_HI (OFF_11_LO + CNT_R)
#define OFF_12_LO (OFF_12_HI + CNT_R)
#define OFF_22_HI (OFF_12_LO + CNT_R)
#define OFF_22_LO (OFF_22_HI + CNT_R)
#define OFF_O_HI  (OFF_22_LO + CNT_R)
#define OFF_O_LO  (OFF_O_HI + CNT_O)

// K-within-32 position held by (lane, element j) for the 16-bit WMMA A operand
// (ISA 7.12.2: lanes 0-15 hold K 0..7 & 16..23; lanes 16-31 hold K 8..15 & 24..31)
__device__ __forceinline__ int klocal(int lane, int j) {
  int v = j >> 1, r = j & 1;
  int base = (lane < 16) ? 0 : 8;
  return (v < 4) ? (base + 2 * v + r) : (16 + base + 2 * (v - 4) + r);
}

__device__ __forceinline__ bf16x16 load_B(const __bf16* __restrict__ W, int tile, int lane) {
  return *(const bf16x16*)(W + (((size_t)tile * 32) + lane) * 16);  // 32B contiguous per lane
}

// Build a 16x32 bf16 A operand from a [16][HID] bf16 LDS spike tile.
__device__ __forceinline__ bf16x16 load_spikes_A(const __bf16* s, int lane, int kt) {
  const int row  = lane & 15;
  const int base = (lane < 16) ? 0 : 8;
  const __bf16* p = s + row * HID + kt * 32;
  bf16x8 r0 = *(const bf16x8*)(p + base);
  bf16x8 r1 = *(const bf16x8*)(p + 16 + base);
  bf16x16 a;
#pragma unroll
  for (int j = 0; j < 8; ++j) { a[j] = r0[j]; a[8 + j] = r1[j]; }
  return a;
}

// ---------------------------------------------------------------------------
// Kernel 1: swizzle weights into per-lane WMMA B-operand layout, bf16 hi + lo.
// ---------------------------------------------------------------------------
__global__ __launch_bounds__(256) void prep_weights(
    const float* __restrict__ Wih1, const float* __restrict__ W11,
    const float* __restrict__ W12,  const float* __restrict__ W22,
    const float* __restrict__ Wo,   __bf16* __restrict__ ws) {
  int gid = blockIdx.x * 256 + threadIdx.x;  // 560*256 == total positions
  const float* src;
  int K, N, NT, idx;
  size_t hi_off, lo_off;
  if (gid < CNT_IH) {
    src = Wih1; K = DIN; N = HID; NT = 8; idx = gid;
    hi_off = OFF_IH_HI; lo_off = OFF_IH_LO;
  } else if (gid < CNT_IH + CNT_R) {
    src = W11; K = HID; N = HID; NT = 8; idx = gid - CNT_IH;
    hi_off = OFF_11_HI; lo_off = OFF_11_LO;
  } else if (gid < CNT_IH + 2 * CNT_R) {
    src = W12; K = HID; N = HID; NT = 8; idx = gid - (CNT_IH + CNT_R);
    hi_off = OFF_12_HI; lo_off = OFF_12_LO;
  } else if (gid < CNT_IH + 3 * CNT_R) {
    src = W22; K = HID; N = HID; NT = 8; idx = gid - (CNT_IH + 2 * CNT_R);
    hi_off = OFF_22_HI; lo_off = OFF_22_LO;
  } else {
    src = Wo; K = HID; N = OUTN; NT = 2; idx = gid - (CNT_IH + 3 * CNT_R);
    hi_off = OFF_O_HI; lo_off = OFF_O_LO;
  }
  int j = idx & 15;
  int lane = (idx >> 4) & 31;
  int tile = idx >> 9;
  int nt = tile % NT;
  int kt = tile / NT;
  int k = kt * 32 + klocal(lane, j);
  int n = nt * 16 + (lane & 15);
  float w = (k < K && n < N) ? src[(size_t)n * K + k] : 0.0f;  // B[k][n] = W[n][k]
  __bf16 hi = (__bf16)w;
  float lo = w - (float)hi;
  ws[hi_off + idx] = hi;
  ws[lo_off + idx] = (__bf16)lo;
}

// ---------------------------------------------------------------------------
// Kernel 2: Xp[t][btile][nt][lane][8] = x[b,t,:] @ W_ih1^T  (bf16x3 emulation)
// 8 waves / block, one 16x128 output tile per wave. 528 WMMAs / wave.
// ---------------------------------------------------------------------------
__global__ __launch_bounds__(256) void xproj_kernel(
    const float* __restrict__ x, const __bf16* __restrict__ Wh,
    const __bf16* __restrict__ Wl, float* __restrict__ Xp) {
  const int lane = threadIdx.x & 31;
  const int wid  = threadIdx.x >> 5;
  const int mt = blockIdx.x * 8 + wid;  // 0..7999
  const int t = mt >> 5;
  const int btile = mt & 31;
  const int row  = lane & 15;
  const int base = (lane < 16) ? 0 : 8;
  const int b = btile * 16 + row;
  const float* __restrict__ xr = x + ((size_t)b * TSTEPS + t) * DIN;

  v8f acc[8];
#pragma unroll
  for (int nt = 0; nt < 8; ++nt)
#pragma unroll
    for (int e = 0; e < 8; ++e) acc[nt][e] = 0.0f;

#pragma unroll 1
  for (int kt = 0; kt < 22; ++kt) {
    __builtin_prefetch(xr + kt * 32 + 96, 0, 1);  // stream input ahead
    bf16x16 ah, al;
#pragma unroll
    for (int j = 0; j < 8; ++j) {
      int k0 = kt * 32 + base + j;
      float x0 = (k0 < DIN) ? xr[k0] : 0.0f;
      __bf16 h0 = (__bf16)x0;
      ah[j] = h0; al[j] = (__bf16)(x0 - (float)h0);
      int k1 = kt * 32 + 16 + base + j;
      float x1 = (k1 < DIN) ? xr[k1] : 0.0f;
      __bf16 h1 = (__bf16)x1;
      ah[8 + j] = h1; al[8 + j] = (__bf16)(x1 - (float)h1);
    }
#pragma unroll
    for (int nt = 0; nt < 8; ++nt) {
      int tile = kt * 8 + nt;
      bf16x16 bh = load_B(Wh, tile, lane);
      bf16x16 bl = load_B(Wl, tile, lane);
      acc[nt] = WMMA_BF16(ah, bh, acc[nt]);  // hi*hi
      acc[nt] = WMMA_BF16(ah, bl, acc[nt]);  // hi*lo
      acc[nt] = WMMA_BF16(al, bh, acc[nt]);  // lo*hi
    }
  }
  float* op = Xp + (size_t)mt * 8 * 256 + lane * 8;
#pragma unroll
  for (int nt = 0; nt < 8; ++nt) *(v8f*)(op + nt * 256) = acc[nt];
}

// ---------------------------------------------------------------------------
// Kernel 3: sequential scan. 32 persistent WGs x 4 waves; WG owns 16 batch
// rows; wave owns 2 N-tiles of hidden state in WMMA C-layout registers.
// Spikes are exchanged via LDS in bf16 (exact). Xp aliases mem_seq[t] slice.
// ---------------------------------------------------------------------------
__global__ __launch_bounds__(128) void scan_kernel(
    const __bf16* __restrict__ W11h, const __bf16* __restrict__ W11l,
    const __bf16* __restrict__ W12h, const __bf16* __restrict__ W12l,
    const __bf16* __restrict__ W22h, const __bf16* __restrict__ W22l,
    const __bf16* __restrict__ Woh,  const __bf16* __restrict__ Wol,
    const float* __restrict__ hid1_mem0, const float* __restrict__ hid2_mem0,
    const float* __restrict__ out_mem0,
    const float* __restrict__ b_ih1,  const float* __restrict__ b_h1h1,
    const float* __restrict__ b_h1h2, const float* __restrict__ b_h2h2,
    const float* __restrict__ b_h2o,
    const float* __restrict__ tau_adp_h1, const float* __restrict__ tau_adp_h2,
    const float* __restrict__ tau_m_h1, const float* __restrict__ tau_m_h2,
    const float* __restrict__ tau_m_o,
    const float* __restrict__ Xp,  // [T][32][8][32][8] (aliases mem_seq)
    float* __restrict__ out_acc, float* __restrict__ spk_seq,
    float* __restrict__ mem_seq, float* __restrict__ omem_seq) {
  const int lane = threadIdx.x & 31;
  const int wid  = threadIdx.x >> 5;  // 0..3
  const int btile = blockIdx.x;       // 0..31
  const int b0 = btile * 16;
  const int half = lane >> 4;
  const int nl = lane & 15;
  const int nt0 = wid * 2;  // this wave's first global N-tile

  __shared__ alignas(32) __bf16 s1_lds[16 * HID];
  __shared__ alignas(32) __bf16 s2_lds[16 * HID];
  for (int i = threadIdx.x; i < 16 * HID; i += 128) {
    s1_lds[i] = (__bf16)0.0f;
    s2_lds[i] = (__bf16)0.0f;
  }
  __syncthreads();

  // Per-neuron constants collapse to per-(lane, q) scalars in C layout.
  float a1[2], r1[2], a2[2], r2[2], bi1[2], bi2[2];
#pragma unroll
  for (int q = 0; q < 2; ++q) {
    int n = (nt0 + q) * 16 + nl;
    a1[q] = expf(-1.0f / tau_m_h1[n]);
    a2[q] = expf(-1.0f / tau_m_h2[n]);
    r1[q] = expf(-1.0f / tau_adp_h1[n]);
    r2[q] = expf(-1.0f / tau_adp_h2[n]);
    bi1[q] = b_ih1[n] + b_h1h1[n];
    bi2[q] = b_h1h2[n] + b_h2h2[n];
  }
  float ao[2], bo[2];
#pragma unroll
  for (int q = 0; q < 2; ++q) {
    int n = q * 16 + nl;
    ao[q] = (n < OUTN) ? expf(-1.0f / tau_m_o[n]) : 0.0f;
    bo[q] = (n < OUTN) ? b_h2o[n] : 0.0f;
  }

  v8f m1[2], b1v[2], s1v[2], m2[2], b2v[2], s2v[2], mo[2], acc[2];
#pragma unroll
  for (int q = 0; q < 2; ++q)
#pragma unroll
    for (int e = 0; e < 8; ++e) {
      int row = e + 8 * half;
      int n = (nt0 + q) * 16 + nl;
      m1[q][e] = hid1_mem0[(size_t)(b0 + row) * HID + n];
      m2[q][e] = hid2_mem0[(size_t)(b0 + row) * HID + n];
      b1v[q][e] = B_J0; b2v[q][e] = B_J0;
      s1v[q][e] = 0.0f; s2v[q][e] = 0.0f;
      int no = q * 16 + nl;
      mo[q][e] = (no < OUTN) ? out_mem0[(size_t)(b0 + row) * OUTN + no] : 0.0f;
      acc[q][e] = 0.0f;
    }

#pragma unroll 1
  for (int t = 0; t < TSTEPS; ++t) {
    // ----- layer 1: i1 = Xp[t] + bias + s1_prev @ W_h1h1^T -----
    v8f i1[2];
    const float* xp = Xp + ((size_t)t * 32 + btile) * 8 * 256;
#pragma unroll
    for (int q = 0; q < 2; ++q) {
      i1[q] = *(const v8f*)(xp + (nt0 + q) * 256 + lane * 8);
#pragma unroll
      for (int e = 0; e < 8; ++e) i1[q][e] += bi1[q];
    }
#pragma unroll
    for (int kt = 0; kt < 4; ++kt) {
      bf16x16 A = load_spikes_A(s1_lds, lane, kt);
#pragma unroll
      for (int q = 0; q < 2; ++q) {
        int tile = kt * 8 + nt0 + q;
        i1[q] = WMMA_BF16(A, load_B(W11h, tile, lane), i1[q]);  // exact: s in {0,1}
        i1[q] = WMMA_BF16(A, load_B(W11l, tile, lane), i1[q]);
      }
    }
#pragma unroll
    for (int q = 0; q < 2; ++q)
#pragma unroll
      for (int e = 0; e < 8; ++e) {
        float sp = s1v[q][e];
        float bb = r1[q] * b1v[q][e] + (1.0f - r1[q]) * sp;
        float Bth = B_J0 + BETA * bb;
        float mm = m1[q][e] * a1[q] + (1.0f - a1[q]) * i1[q][e] - Bth * sp;
        b1v[q][e] = bb;
        m1[q][e] = mm;
        s1v[q][e] = (mm - Bth > 0.0f) ? 1.0f : 0.0f;
      }
    __syncthreads();  // all waves done reading s1_prev
#pragma unroll
    for (int q = 0; q < 2; ++q)
#pragma unroll
      for (int e = 0; e < 8; ++e)
        s1_lds[(e + 8 * half) * HID + (nt0 + q) * 16 + nl] = (__bf16)s1v[q][e];
    __syncthreads();

    // ----- layer 2: i2 = s1_new @ W_h1h2^T + s2_prev @ W_h2h2^T + bias -----
    v8f i2[2];
#pragma unroll
    for (int q = 0; q < 2; ++q)
#pragma unroll
      for (int e = 0; e < 8; ++e) i2[q][e] = bi2[q];
#pragma unroll
    for (int kt = 0; kt < 4; ++kt) {
      bf16x16 A1 = load_spikes_A(s1_lds, lane, kt);
      bf16x16 A2 = load_spikes_A(s2_lds, lane, kt);
#pragma unroll
      for (int q = 0; q < 2; ++q) {
        int tile = kt * 8 + nt0 + q;
        i2[q] = WMMA_BF16(A1, load_B(W12h, tile, lane), i2[q]);
        i2[q] = WMMA_BF16(A1, load_B(W12l, tile, lane), i2[q]);
        i2[q] = WMMA_BF16(A2, load_B(W22h, tile, lane), i2[q]);
        i2[q] = WMMA_BF16(A2, load_B(W22l, tile, lane), i2[q]);
      }
    }
#pragma unroll
    for (int q = 0; q < 2; ++q)
#pragma unroll
      for (int e = 0; e < 8; ++e) {
        float sp = s2v[q][e];
        float bb = r2[q] * b2v[q][e] + (1.0f - r2[q]) * sp;
        float Bth = B_J0 + BETA * bb;
        float mm = m2[q][e] * a2[q] + (1.0f - a2[q]) * i2[q][e] - Bth * sp;
        b2v[q][e] = bb;
        m2[q][e] = mm;
        float sn = (mm - Bth > 0.0f) ? 1.0f : 0.0f;
        s2v[q][e] = sn;
        int row = e + 8 * half;
        size_t o = ((size_t)t * BATCH + b0 + row) * HID + (nt0 + q) * 16 + nl;
        spk_seq[o] = sn;
        mem_seq[o] = mm;  // overwrites this WG's consumed Xp chunk only
      }
    __syncthreads();  // all waves done reading s2_prev
#pragma unroll
    for (int q = 0; q < 2; ++q)
#pragma unroll
      for (int e = 0; e < 8; ++e)
        s2_lds[(e + 8 * half) * HID + (nt0 + q) * 16 + nl] = (__bf16)s2v[q][e];
    __syncthreads();

    // ----- readout + softmax accumulation (wave 0) -----
    if (wid == 0) {
      v8f io[2];
#pragma unroll
      for (int q = 0; q < 2; ++q)
#pragma unroll
        for (int e = 0; e < 8; ++e) io[q][e] = bo[q];
#pragma unroll
      for (int kt = 0; kt < 4; ++kt) {
        bf16x16 A = load_spikes_A(s2_lds, lane, kt);
#pragma unroll
        for (int q = 0; q < 2; ++q) {
          int tile = kt * 2 + q;
          io[q] = WMMA_BF16(A, load_B(Woh, tile, lane), io[q]);
          io[q] = WMMA_BF16(A, load_B(Wol, tile, lane), io[q]);
        }
      }
#pragma unroll
      for (int q = 0; q < 2; ++q)
#pragma unroll
        for (int e = 0; e < 8; ++e)
          mo[q][e] = mo[q][e] * ao[q] + (1.0f - ao[q]) * io[q][e];
      float gate = (t > 10) ? 1.0f : 0.0f;
#pragma unroll
      for (int e = 0; e < 8; ++e) {
        int row = e + 8 * half;
        size_t ob = ((size_t)t * BATCH + b0 + row) * OUTN;
        omem_seq[ob + nl] = mo[0][e];
        if (nl < OUTN - 16) omem_seq[ob + 16 + nl] = mo[1][e];
        // row-wise softmax over 20 cols: cols live across 16-lane group (+2nd tile)
        float v0 = mo[0][e];
        float v1 = (nl < OUTN - 16) ? mo[1][e] : -3.0e38f;
        float mx = fmaxf(v0, v1);
#pragma unroll
        for (int m = 1; m < 16; m <<= 1) mx = fmaxf(mx, __shfl_xor(mx, m, 32));
        float p0 = expf(v0 - mx);
        float p1 = (nl < OUTN - 16) ? expf(v1 - mx) : 0.0f;
        float sm = p0 + p1;
#pragma unroll
        for (int m = 1; m < 16; m <<= 1) sm += __shfl_xor(sm, m, 32);
        float ginv = gate / sm;
        acc[0][e] += p0 * ginv;
        acc[1][e] += p1 * ginv;
      }
    }
  }

  if (wid == 0) {
#pragma unroll
    for (int e = 0; e < 8; ++e) {
      int row = e + 8 * half;
      out_acc[(size_t)(b0 + row) * OUTN + nl] = acc[0][e];
      if (nl < OUTN - 16) out_acc[(size_t)(b0 + row) * OUTN + 16 + nl] = acc[1][e];
    }
  }
}

// ---------------------------------------------------------------------------
extern "C" void kernel_launch(void* const* d_in, const int* in_sizes, int n_in,
                              void* d_out, int out_size, void* d_ws, size_t ws_size,
                              hipStream_t stream) {
  (void)in_sizes; (void)n_in; (void)out_size; (void)ws_size;
  const float* input      = (const float*)d_in[0];
  const float* hid1_mem0  = (const float*)d_in[3];
  const float* hid2_mem0  = (const float*)d_in[4];
  const float* out_mem0   = (const float*)d_in[5];
  const float* W_ih1      = (const float*)d_in[6];
  const float* b_ih1      = (const float*)d_in[7];
  const float* W_h1h1     = (const float*)d_in[8];
  const float* b_h1h1     = (const float*)d_in[9];
  const float* W_h1h2     = (const float*)d_in[10];
  const float* b_h1h2     = (const float*)d_in[11];
  const float* W_h2h2     = (const float*)d_in[12];
  const float* b_h2h2     = (const float*)d_in[13];
  const float* W_h2o      = (const float*)d_in[14];
  const float* b_h2o      = (const float*)d_in[15];
  const float* tau_adp_h1 = (const float*)d_in[16];
  const float* tau_adp_h2 = (const float*)d_in[17];
  const float* tau_m_h1   = (const float*)d_in[18];
  const float* tau_m_h2   = (const float*)d_in[19];
  const float* tau_m_o    = (const float*)d_in[20];

  float* out  = (float*)d_out;                           // [512][20]
  float* spk  = out + (size_t)BATCH * OUTN;              // [250][512][128]
  float* mem  = spk + (size_t)TSTEPS * BATCH * HID;      // [250][512][128] (Xp alias)
  float* omem = mem + (size_t)TSTEPS * BATCH * HID;      // [250][512][20]
  __bf16* ws = (__bf16*)d_ws;

  prep_weights<<<560, 256, 0, stream>>>(W_ih1, W_h1h1, W_h1h2, W_h2h2, W_h2o, ws);
  xproj_kernel<<<1000, 256, 0, stream>>>(input, ws + OFF_IH_HI, ws + OFF_IH_LO, mem);
  scan_kernel<<<32, 128, 0, stream>>>(
      ws + OFF_11_HI, ws + OFF_11_LO, ws + OFF_12_HI, ws + OFF_12_LO,
      ws + OFF_22_HI, ws + OFF_22_LO, ws + OFF_O_HI, ws + OFF_O_LO,
      hid1_mem0, hid2_mem0, out_mem0,
      b_ih1, b_h1h1, b_h1h2, b_h2h2, b_h2o,
      tau_adp_h1, tau_adp_h2, tau_m_h1, tau_m_h2, tau_m_o,
      mem, out, spk, mem, omem);
}
